// CrossAttention_11175504904849
// MI455X (gfx1250) — matrix operand order repaired and checked by reading, not compiled, not run on hardware
//
#include <hip/hip_runtime.h>
#include <hip/hip_bf16.h>

#define NHID 512
#define PROJ 64
#define NC   4096
#define NE   8192
#define NG   64

#define POOL_CHUNK 256
#define NPB (NC / POOL_CHUNK)   // 16 partial blocks

typedef __bf16 bf16_t;
typedef __attribute__((ext_vector_type(16))) __bf16 v16bf;
typedef __attribute__((ext_vector_type(8)))  float  v8f;
typedef __attribute__((ext_vector_type(4)))  unsigned int v4u;

union ABu { v16bf v; v4u q[2]; };

__device__ __forceinline__ v8f v8f_zero() {
    v8f z = {0.f,0.f,0.f,0.f,0.f,0.f,0.f,0.f};
    return z;
}

__device__ __forceinline__ v8f wmma_bf16(v16bf a, v16bf b, v8f c) {
    return __builtin_amdgcn_wmma_f32_16x16x32_bf16(false, a, false, b, (short)0, c, false, false);
}

// A operand (16x32 bf16): lane<16 -> M=lane, K chunks {kb..kb+7},{kb+16..kb+23};
// lane>=16 -> M=lane-16, chunks shifted by 8.
__device__ __forceinline__ v16bf load_A16(const bf16_t* rowbase, int kbase, int lane) {
    ABu r;
    int o1 = kbase + ((lane >> 4) << 3);
    r.q[0] = *(const v4u*)(rowbase + o1);
    r.q[1] = *(const v4u*)(rowbase + o1 + 16);
    return r.v;
}

// B operand (32x16 bf16): lane%16 = N column; lane<16 -> K chunks {kb..kb+7},{kb+16..kb+23}; lane>=16 -> chunks shifted by 8
__device__ __forceinline__ v16bf load_B16(const bf16_t* rowbase, int kbase, int lane) {
    ABu r;
    int o = kbase + ((lane >> 4) << 3);
    r.q[0] = *(const v4u*)(rowbase + o);
    r.q[1] = *(const v4u*)(rowbase + o + 16);
    return r.v;
}

struct f8 { float4 lo, hi; };
__device__ __forceinline__ f8 ld8(const float* p) {
    f8 r; r.lo = ((const float4*)p)[0]; r.hi = ((const float4*)p)[1]; return r;
}

// Convert fp32 row to bf16 A operand on the fly
__device__ __forceinline__ v16bf cvtA_f32(const float* rowbase, int kbase, int lane) {
    int o1 = kbase + ((lane >> 4) << 3);
    f8 a = ld8(rowbase + o1);
    f8 b = ld8(rowbase + o1 + 16);
    v16bf v;
    v[0]=(bf16_t)a.lo.x;  v[1]=(bf16_t)a.lo.y;  v[2]=(bf16_t)a.lo.z;  v[3]=(bf16_t)a.lo.w;
    v[4]=(bf16_t)a.hi.x;  v[5]=(bf16_t)a.hi.y;  v[6]=(bf16_t)a.hi.z;  v[7]=(bf16_t)a.hi.w;
    v[8]=(bf16_t)b.lo.x;  v[9]=(bf16_t)b.lo.y;  v[10]=(bf16_t)b.lo.z; v[11]=(bf16_t)b.lo.w;
    v[12]=(bf16_t)b.hi.x; v[13]=(bf16_t)b.hi.y; v[14]=(bf16_t)b.hi.z; v[15]=(bf16_t)b.hi.w;
    return v;
}

// ---------------- prep: weight transposes, evidence^T gather, batch gathers ----------------
#define SEG0 (64*512)        // WcT
#define SEG1 (64*512)        // WeT
#define SEG2 (512*2048)      // WaT
#define SEG3 (512*8192)      // evT
#define SEG4 (NC)            // cb
#define SEG5 (NE)            // eb
#define PREP_TOTAL (SEG0+SEG1+SEG2+SEG3+SEG4+SEG5)

__global__ void k_prep(const float* __restrict__ x, const int* __restrict__ batch,
                       const int* __restrict__ cidx, const int* __restrict__ eidx,
                       const float* __restrict__ Wc, const float* __restrict__ We,
                       const float* __restrict__ Wa,
                       bf16_t* __restrict__ WcT, bf16_t* __restrict__ WeT,
                       bf16_t* __restrict__ WaT, bf16_t* __restrict__ evT,
                       int* __restrict__ cb, int* __restrict__ eb) {
    long id = (long)blockIdx.x * blockDim.x + threadIdx.x;
    if (id < SEG0) {
        long q = id; int n = (int)(q / 512); int k = (int)(q % 512);
        WcT[q] = (bf16_t)Wc[(long)k * 64 + n];
        return;
    }
    id -= SEG0;
    if (id < SEG1) {
        long q = id; int n = (int)(q / 512); int k = (int)(q % 512);
        WeT[q] = (bf16_t)We[(long)k * 64 + n];
        return;
    }
    id -= SEG1;
    if (id < SEG2) {
        long q = id; int n = (int)(q / 2048); int k = (int)(q % 2048);
        WaT[q] = (bf16_t)Wa[(long)k * 512 + n];
        return;
    }
    id -= SEG2;
    if (id < SEG3) {
        long q = id; int j = (int)(q % NE); int k = (int)(q / NE);
        evT[q] = (bf16_t)x[(long)eidx[j] * NHID + k];
        return;
    }
    id -= SEG3;
    if (id < SEG4) { cb[id] = batch[cidx[id]]; return; }
    id -= SEG4;
    if (id < SEG5) { eb[id] = batch[eidx[id]]; }
}

// ---------------- projection: out[M,64] = bf16( x[idx] @ W^T-stored + bias ) ---------------
__global__ void k_proj(const float* __restrict__ x, const int* __restrict__ idx,
                       const bf16_t* __restrict__ WT, const float* __restrict__ bias,
                       bf16_t* __restrict__ out) {
    int lane = threadIdx.x & 31;
    int m0 = blockIdx.x * 16;
    const float* xrow = x + (long)idx[m0 + (lane & 15)] * NHID;
    v8f acc[4];
    #pragma unroll
    for (int t = 0; t < 4; ++t) acc[t] = v8f_zero();
    for (int ks = 0; ks < NHID; ks += 32) {
        v16bf a = cvtA_f32(xrow, ks, lane);
        #pragma unroll
        for (int t = 0; t < 4; ++t) {
            const bf16_t* brow = WT + (long)(16*t + (lane & 15)) * NHID;
            acc[t] = wmma_bf16(a, load_B16(brow, ks, lane), acc[t]);
        }
    }
    #pragma unroll
    for (int t = 0; t < 4; ++t) {
        int col = 16*t + (lane & 15);
        float bz = bias[col];
        #pragma unroll
        for (int r = 0; r < 8; ++r) {
            int m = m0 + r + ((lane >> 4) << 3);
            out[(long)m * PROJ + col] = (bf16_t)(acc[t][r] + bz);
        }
    }
}

// ---------------- softmax row statistics (max, sumexp) over masked scores ------------------
__global__ void k_stats(const bf16_t* __restrict__ wq, const bf16_t* __restrict__ wk,
                        const int* __restrict__ cb, const int* __restrict__ eb,
                        float* __restrict__ m_arr, float* __restrict__ l_arr) {
    int lane = threadIdx.x & 31;
    int m0 = blockIdx.x * 16;
    const bf16_t* arow = wq + (long)(m0 + (lane & 15)) * PROJ;
    v16bf Aq0 = load_A16(arow, 0, lane);
    v16bf Aq1 = load_A16(arow, 32, lane);
    int cbr[8];
    #pragma unroll
    for (int r = 0; r < 8; ++r) cbr[r] = cb[m0 + r + ((lane >> 4) << 3)];

    float vmax[8];
    #pragma unroll
    for (int r = 0; r < 8; ++r) vmax[r] = -__builtin_inff();
    for (int j = 0; j < NE; j += 16) {
        const bf16_t* brow = wk + (long)(j + (lane & 15)) * PROJ;
        v8f c = v8f_zero();
        c = wmma_bf16(Aq0, load_B16(brow, 0, lane), c);
        c = wmma_bf16(Aq1, load_B16(brow, 32, lane), c);
        int ebn = eb[j + (lane & 15)];
        #pragma unroll
        for (int r = 0; r < 8; ++r)
            if (cbr[r] == ebn) vmax[r] = fmaxf(vmax[r], c[r]);
    }
    #pragma unroll
    for (int r = 0; r < 8; ++r)
        for (int off = 1; off < 16; off <<= 1)
            vmax[r] = fmaxf(vmax[r], __shfl_xor(vmax[r], off, 32));

    float vsum[8];
    #pragma unroll
    for (int r = 0; r < 8; ++r) vsum[r] = 0.f;
    for (int j = 0; j < NE; j += 16) {
        const bf16_t* brow = wk + (long)(j + (lane & 15)) * PROJ;
        v8f c = v8f_zero();
        c = wmma_bf16(Aq0, load_B16(brow, 0, lane), c);
        c = wmma_bf16(Aq1, load_B16(brow, 32, lane), c);
        int ebn = eb[j + (lane & 15)];
        #pragma unroll
        for (int r = 0; r < 8; ++r)
            if (cbr[r] == ebn) vsum[r] += __expf(c[r] - vmax[r]);
    }
    #pragma unroll
    for (int r = 0; r < 8; ++r)
        for (int off = 1; off < 16; off <<= 1)
            vsum[r] += __shfl_xor(vsum[r], off, 32);

    if ((lane & 15) == 0) {
        #pragma unroll
        for (int r = 0; r < 8; ++r) {
            int m = m0 + r + ((lane >> 4) << 3);
            m_arr[m] = vmax[r];
            l_arr[m] = vsum[r];
        }
    }
}

// ---------------- attention * V: claim_new = softmax(scores) @ evidence --------------------
__global__ void k_av(const bf16_t* __restrict__ wq, const bf16_t* __restrict__ wk,
                     const bf16_t* __restrict__ evT,
                     const int* __restrict__ cb, const int* __restrict__ eb,
                     const float* __restrict__ m_arr, const float* __restrict__ l_arr,
                     float* __restrict__ cn) {
    int tid = threadIdx.x, lane = tid & 31, w = tid >> 5;
    int m0 = blockIdx.x * 16;
    __shared__ bf16_t P[16 * 256];   // probability tile [claim row][local evidence]

    const bf16_t* arow = wq + (long)(m0 + (lane & 15)) * PROJ;
    v16bf Aq0 = load_A16(arow, 0, lane);
    v16bf Aq1 = load_A16(arow, 32, lane);

    int cbr[8]; float mv[8], li[8];
    #pragma unroll
    for (int r = 0; r < 8; ++r) {
        int m = m0 + r + ((lane >> 4) << 3);
        cbr[r] = cb[m]; mv[r] = m_arr[m];
        li[r] = 1.0f / fmaxf(l_arr[m], 1e-30f);
    }

    v8f acc[4];
    #pragma unroll
    for (int t = 0; t < 4; ++t) acc[t] = v8f_zero();

    for (int e0 = 0; e0 < NE; e0 += 256) {
        if (e0 + 256 < NE) {
            // warm the next evidence chunk (K operand rows) into cache
            __builtin_prefetch(wk + (long)(e0 + 256 + 32*w + (lane & 31)) * PROJ, 0, 1);
        }
        #pragma unroll
        for (int t = 0; t < 2; ++t) {
            int ebase = e0 + 32*w + 16*t;
            const bf16_t* brow = wk + (long)(ebase + (lane & 15)) * PROJ;
            v8f c = v8f_zero();
            c = wmma_bf16(Aq0, load_B16(brow, 0, lane), c);
            c = wmma_bf16(Aq1, load_B16(brow, 32, lane), c);
            int ebn = eb[ebase + (lane & 15)];
            int kloc = 32*w + 16*t + (lane & 15);
            #pragma unroll
            for (int r = 0; r < 8; ++r) {
                float p = (cbr[r] == ebn) ? __expf(c[r] - mv[r]) * li[r] : 0.0f;
                P[(r + ((lane >> 4) << 3)) * 256 + kloc] = (bf16_t)p;
            }
        }
        __syncthreads();
        #pragma unroll
        for (int ksub = 0; ksub < 8; ++ksub) {
            v16bf a = load_A16(&P[(lane & 15) * 256], ksub * 32, lane);
            #pragma unroll
            for (int t2 = 0; t2 < 4; ++t2) {
                const bf16_t* brow = evT + (long)(64*w + 16*t2 + (lane & 15)) * NE
                                        + e0 + ksub * 32;
                acc[t2] = wmma_bf16(a, load_B16(brow, 0, lane), acc[t2]);
            }
        }
        __syncthreads();
    }

    #pragma unroll
    for (int t2 = 0; t2 < 4; ++t2) {
        int col = 64*w + 16*t2 + (lane & 15);
        #pragma unroll
        for (int r = 0; r < 8; ++r)
            cn[(long)(m0 + r + ((lane >> 4) << 3)) * NHID + col] = acc[t2][r];
    }
}

// ---------------- output GEMM: a = [c, c', c-c', c*c'] @ Wa + ba ---------------------------
__global__ void k_out(const float* __restrict__ x, const int* __restrict__ cidx,
                      const float* __restrict__ cn, const bf16_t* __restrict__ WaT,
                      const float* __restrict__ ba, float* __restrict__ abuf) {
    int tid = threadIdx.x, lane = tid & 31, w = tid >> 5;
    int m0 = blockIdx.x * 16;
    int mrow = m0 + (lane & 15);
    const float* xr = x + (long)cidx[mrow] * NHID;
    const float* nr = cn + (long)mrow * NHID;

    v8f acc[4];
    #pragma unroll
    for (int t = 0; t < 4; ++t) acc[t] = v8f_zero();

    for (int ks = 0; ks < 4 * NHID; ks += 32) {
        int sec = ks >> 9;
        int kk  = ks & 511;
        int o1  = kk + ((lane >> 4) << 3);
        f8 c0 = ld8(xr + o1), c1 = ld8(xr + o1 + 16);
        f8 n0 = ld8(nr + o1), n1 = ld8(nr + o1 + 16);
        float cf[16] = {c0.lo.x,c0.lo.y,c0.lo.z,c0.lo.w, c0.hi.x,c0.hi.y,c0.hi.z,c0.hi.w,
                        c1.lo.x,c1.lo.y,c1.lo.z,c1.lo.w, c1.hi.x,c1.hi.y,c1.hi.z,c1.hi.w};
        float nf[16] = {n0.lo.x,n0.lo.y,n0.lo.z,n0.lo.w, n0.hi.x,n0.hi.y,n0.hi.z,n0.hi.w,
                        n1.lo.x,n1.lo.y,n1.lo.z,n1.lo.w, n1.hi.x,n1.hi.y,n1.hi.z,n1.hi.w};
        v16bf a;
        #pragma unroll
        for (int i = 0; i < 16; ++i) {
            float cv = cf[i], nv = nf[i];
            float r = (sec == 0) ? cv : (sec == 1) ? nv : (sec == 2) ? (cv - nv) : (cv * nv);
            a[i] = (bf16_t)r;
        }
        #pragma unroll
        for (int t = 0; t < 4; ++t) {
            const bf16_t* brow = WaT + (long)(64*w + 16*t + (lane & 15)) * (4 * NHID);
            acc[t] = wmma_bf16(a, load_B16(brow, ks, lane), acc[t]);
        }
    }
    #pragma unroll
    for (int t = 0; t < 4; ++t) {
        int col = 64*w + 16*t + (lane & 15);
        float bz = ba[col];
        #pragma unroll
        for (int r = 0; r < 8; ++r)
            abuf[(long)(m0 + r + ((lane >> 4) << 3)) * NHID + col] = acc[t][r] + bz;
    }
}

// ---------------- deterministic two-stage segment-mean pool --------------------------------
// Stage 1: each block reduces POOL_CHUNK rows into a [NG][NHID] f32 LDS accumulator
// (128 KB of the WGP's 320 KB LDS) and writes one partial + per-graph counts.
__global__ void k_pool1(const float* __restrict__ abuf, const int* __restrict__ cb,
                        float* __restrict__ partial, float* __restrict__ pcnt) {
    __shared__ float acc[NG * NHID];           // 128 KB
    __shared__ float cnts[NG];
    int tid = threadIdx.x;                     // 256 threads
    for (int i = tid; i < NG * NHID; i += 256) acc[i] = 0.f;
    if (tid < NG) cnts[tid] = 0.f;
    __syncthreads();

    int m0 = blockIdx.x * POOL_CHUNK;
    for (int mm = 0; mm < POOL_CHUNK; ++mm) {
        int m = m0 + mm;
        int g = cb[m];
        const float* row = abuf + (long)m * NHID;
        acc[g * NHID + tid]        += row[tid];
        acc[g * NHID + tid + 256]  += row[tid + 256];
        if (tid == 0) cnts[g] += 1.f;
    }
    __syncthreads();

    float* pout = partial + (long)blockIdx.x * NG * NHID;
    for (int i = tid; i < NG * NHID; i += 256) pout[i] = acc[i];
    if (tid < NG) pcnt[blockIdx.x * NG + tid] = cnts[tid];
}

// Stage 2: fold NPB partials, divide by counts.
__global__ void k_pool2(const float* __restrict__ partial, const float* __restrict__ pcnt,
                        float* __restrict__ out) {
    int g = blockIdx.x;
    int c = threadIdx.x;
    float s = 0.f, cnt = 0.f;
    #pragma unroll
    for (int b = 0; b < NPB; ++b) {
        s   += partial[((long)b * NG + g) * NHID + c];
        cnt += pcnt[b * NG + g];
    }
    out[(long)g * NHID + c] = s / fmaxf(cnt, 1.0f);
}

extern "C" void kernel_launch(void* const* d_in, const int* in_sizes, int n_in,
                              void* d_out, int out_size, void* d_ws, size_t ws_size,
                              hipStream_t stream) {
    (void)in_sizes; (void)n_in; (void)out_size; (void)ws_size;
    const float* x     = (const float*)d_in[0];
    const int*   batch = (const int*)d_in[1];
    const int*   cidx  = (const int*)d_in[2];
    const int*   eidx  = (const int*)d_in[3];
    const float* Wc    = (const float*)d_in[4];
    const float* bc    = (const float*)d_in[5];
    const float* We    = (const float*)d_in[6];
    const float* be    = (const float*)d_in[7];
    const float* Wa    = (const float*)d_in[8];
    const float* ba    = (const float*)d_in[9];
    float* out = (float*)d_out;

    char* base = (char*)d_ws;
    size_t off = 0;
    auto alloc = [&](size_t bytes) -> char* {
        char* p = base + off;
        off = (off + bytes + 255) & ~(size_t)255;
        return p;
    };
    bf16_t* evT  = (bf16_t*)alloc((size_t)512 * NE * 2);
    bf16_t* WcT  = (bf16_t*)alloc((size_t)64 * 512 * 2);
    bf16_t* WeT  = (bf16_t*)alloc((size_t)64 * 512 * 2);
    bf16_t* WaT  = (bf16_t*)alloc((size_t)512 * 2048 * 2);
    bf16_t* wq   = (bf16_t*)alloc((size_t)NC * PROJ * 2);
    bf16_t* wk   = (bf16_t*)alloc((size_t)NE * PROJ * 2);
    int*    cb   = (int*)alloc((size_t)NC * 4);
    int*    eb   = (int*)alloc((size_t)NE * 4);
    float*  marr = (float*)alloc((size_t)NC * 4);
    float*  larr = (float*)alloc((size_t)NC * 4);
    float*  cn   = (float*)alloc((size_t)NC * NHID * 4);
    float*  abuf = (float*)alloc((size_t)NC * NHID * 4);
    float*  part = (float*)alloc((size_t)NPB * NG * NHID * 4);
    float*  pcnt = (float*)alloc((size_t)NPB * NG * 4);

    k_prep<<<PREP_TOTAL / 256, 256, 0, stream>>>(x, batch, cidx, eidx, Wc, We, Wa,
                                                 WcT, WeT, WaT, evT, cb, eb);
    k_proj<<<NC / 16, 32, 0, stream>>>(x, cidx, WcT, bc, wq);
    k_proj<<<NE / 16, 32, 0, stream>>>(x, eidx, WeT, be, wk);
    k_stats<<<NC / 16, 32, 0, stream>>>(wq, wk, cb, eb, marr, larr);
    k_av<<<NC / 16, 256, 0, stream>>>(wq, wk, evT, cb, eb, marr, larr, cn);
    k_out<<<NC / 16, 256, 0, stream>>>(x, cidx, cn, WaT, ba, abuf);
    k_pool1<<<NPB, 256, 0, stream>>>(abuf, cb, part, pcnt);
    k_pool2<<<NG, NHID, 0, stream>>>(part, pcnt, out);
}
